// WaveNet_14439680049375
// MI455X (gfx1250) — matrix operand order repaired and checked by reading, not compiled, not run on hardware
//
#include <hip/hip_runtime.h>

// ---------------------------------------------------------------------------
// WaveNet on MI455X (gfx1250, wave32, WMMA f32_16x16x32_f16)
//
// Working set (weights 3.5MB, acts 2MB, G 36MB) is L2-resident (192MB):
// the problem is WMMA-throughput + serial-chain-latency bound.
// skip path is re-associated into one [512 x 1600] x [1600 x 11269] GEMM.
// ---------------------------------------------------------------------------

#define MU        256
#define NRES      32
#define NSKIP     512
#define T_IN      16384
#define NBLOCKS   50
#define FINAL_LEN 11269
#define GSTR      11280   // padded column stride for G buffer (mult of 16)
#define XSTR      16384   // fixed row stride for activation buffers

typedef __attribute__((ext_vector_type(16))) _Float16 v16h;
typedef __attribute__((ext_vector_type(8)))  float    v8f;

static __device__ __forceinline__ v8f wmma_f16(v16h a, v16h b, v8f c) {
  // (neg_a, A, neg_b, B, c_mod, C, reuse_a, reuse_b)
  return __builtin_amdgcn_wmma_f32_16x16x32_f16(false, a, false, b, (short)0, c,
                                                false, false);
}

static __device__ __forceinline__ v8f zero8() {
  v8f z;
#pragma unroll
  for (int i = 0; i < 8; ++i) z[i] = 0.0f;
  return z;
}

// Branch-free fast transcendentals (native v_tanh_f32 on gfx1250 if exposed)
static __device__ __forceinline__ float fast_tanh(float x) {
#if __has_builtin(__builtin_amdgcn_tanhf)
  return __builtin_amdgcn_tanhf(x);
#else
  float e = __expf(2.0f * x);                         // v_exp_f32
  return 1.0f - 2.0f * __builtin_amdgcn_rcpf(e + 1.0f);  // v_rcp_f32
#endif
}
static __device__ __forceinline__ float fast_sigmoid(float x) {
  return __builtin_amdgcn_rcpf(1.0f + __expf(-x));
}
static __device__ __forceinline__ float elu(float x) {
  return x > 0.0f ? x : (__expf(x) - 1.0f);
}

// SWAPX16: exchange lane <-> lane^16 (group-of-32 ds_swizzle, xor=0x10, and=0x1f)
static __device__ __forceinline__ float swap16(float x) {
  return __int_as_float(__builtin_amdgcn_ds_swizzle(__float_as_int(x), 0x401F));
}

// Build 32x16 f16 B fragment (K=channel, N=time) from f32 activations
// x[c*stride + t].  B layout: lane<16 holds K=0..15 of col=lane,
// lane>=16 holds K=16..31 of col=lane-16 (halves in ascending K).
static __device__ __forceinline__ v16h load_bfrag_f32(const float* __restrict__ x,
                                                      int stride, int t0, int tmax) {
  int l = threadIdx.x & 31;
  int t = t0 + (l & 15);
  if (t > tmax) t = tmax;               // clamp tail (columns guarded at store)
  int cbase = (l >> 4) << 4;
  v16h b;
#pragma unroll
  for (int j = 0; j < 16; ++j)
    b[j] = (_Float16)x[(cbase + j) * stride + t];
  return b;
}

// Convert two 16x16 f32 D tiles (rows 0-15 in lo, rows 16-31 in hi) into a
// 32x16 B fragment.  Only needs a lane<->lane^16 exchange (ds_swizzle).
static __device__ __forceinline__ v16h dpair_to_bfrag(v8f lo, v8f hi) {
  int l = threadIdx.x & 31;
  bool low = l < 16;
  v16h b;
#pragma unroll
  for (int v = 0; v < 8; ++v) {
    float sl = swap16(lo[v]);
    float sh = swap16(hi[v]);
    b[v]     = (_Float16)(low ? lo[v] : sh);     // K = v      | 16+v
    b[v + 8] = (_Float16)(low ? sl    : hi[v]);  // K = 8+v    | 24+v
  }
  return b;
}

// ---------------------------------------------------------------------------
// Prep: pack all weight matrices into A-fragment layout (f32 -> f16).
// A layout 16x32: lane<16 row=lane, halves = K{0..15};
//                 lane>=16 row=lane-16, halves = K{16..31}.
// One wave packs one 16x32 fragment (512 halves = 1KB).
// ---------------------------------------------------------------------------
#define NSK_FRAGS (32 * NBLOCKS)   // w_skip concat: 32 Mtiles x 50 ktiles
#define NP1_FRAGS (32 * 16)        // w_p1: 32 Mtiles x 16 ktiles
#define NP2_FRAGS (16 * 16)        // w_p2: 16 Mtiles x 16 ktiles
#define NGT_FRAGS (NBLOCKS * 10)   // per block: sig(4) tanh(4) res(2)
#define TOTAL_FRAGS (NSK_FRAGS + NP1_FRAGS + NP2_FRAGS + NGT_FRAGS)

__global__ __launch_bounds__(128) void pack_kernel(
    const float* __restrict__ w_sig, const float* __restrict__ w_tanh,
    const float* __restrict__ w_skip, const float* __restrict__ w_res,
    const float* __restrict__ w_p1, const float* __restrict__ w_p2,
    _Float16* __restrict__ Wskip, _Float16* __restrict__ Wp1,
    _Float16* __restrict__ Wp2, _Float16* __restrict__ Wgate) {
  int gw = (blockIdx.x * blockDim.x + threadIdx.x) >> 5;
  if (gw >= TOTAL_FRAGS) return;
  int l   = threadIdx.x & 31;
  int rlo = l & 15;
  int khi = (l >> 4) * 16;
  _Float16 vals[16];
  _Float16* dst = nullptr;

  if (gw < NSK_FRAGS) {
    // big skip matrix A[m][k], k = 32*block + c
    int mt = gw / NBLOCKS, kt = gw % NBLOCKS;
    int r = mt * 16 + rlo;
#pragma unroll
    for (int j = 0; j < 16; ++j) {
      int c = j + khi;
      vals[j] = (_Float16)w_skip[((size_t)kt * NSKIP + r) * NRES + c];
    }
    dst = Wskip + (size_t)gw * 512;
  } else if (gw < NSK_FRAGS + NP1_FRAGS) {
    int f = gw - NSK_FRAGS;
    int mt = f / 16, kt = f % 16;
    int r = mt * 16 + rlo;
#pragma unroll
    for (int j = 0; j < 16; ++j) {
      int k = kt * 32 + j + khi;
      vals[j] = (_Float16)w_p1[(size_t)r * NSKIP + k];
    }
    dst = Wp1 + (size_t)f * 512;
  } else if (gw < NSK_FRAGS + NP1_FRAGS + NP2_FRAGS) {
    int f = gw - NSK_FRAGS - NP1_FRAGS;
    int mt = f / 16, kt = f % 16;
    int r = mt * 16 + rlo;
#pragma unroll
    for (int j = 0; j < 16; ++j) {
      int k = kt * 32 + j + khi;
      vals[j] = (_Float16)w_p2[(size_t)r * NSKIP + k];
    }
    dst = Wp2 + (size_t)f * 512;
  } else {
    int f = gw - NSK_FRAGS - NP1_FRAGS - NP2_FRAGS;
    int blk = f / 10, fx = f % 10;
#pragma unroll
    for (int j = 0; j < 16; ++j) {
      int c = j + khi;
      float v;
      if (fx < 8) {  // f = grp*4 + tap*2 + mtile  (grp: 0=sig 1=tanh)
        int grp = fx >> 2, tap = (fx >> 1) & 1, mt = fx & 1;
        int r = mt * 16 + rlo;
        const float* W = grp ? w_tanh : w_sig;
        v = W[(((size_t)blk * NRES + r) * NRES + c) * 2 + tap];
      } else {       // res: fx-8 = mtile
        int r = (fx - 8) * 16 + rlo;
        v = w_res[((size_t)blk * NRES + r) * NRES + c];
      }
      vals[j] = (_Float16)v;
    }
    dst = Wgate + (size_t)f * 512;
  }
  v16h pv;
#pragma unroll
  for (int j = 0; j < 16; ++j) pv[j] = vals[j];
  *(v16h*)(dst + l * 16) = pv;
}

__global__ void bsum_kernel(const float* __restrict__ b_skip,
                            float* __restrict__ Bsum) {
  int m = blockIdx.x * blockDim.x + threadIdx.x;
  if (m < NSKIP) {
    float s = 0.f;
    for (int i = 0; i < NBLOCKS; ++i) s += b_skip[i * NSKIP + m];
    Bsum[m] = s;
  }
}

// one-hot @ w_in == gather
__global__ void embed_kernel(const int* __restrict__ idx,
                             const float* __restrict__ w_in,
                             const float* __restrict__ b_in,
                             float* __restrict__ x0) {
  int t = blockIdx.x * blockDim.x + threadIdx.x;
  if (t >= T_IN) return;
  int id = idx[t];
#pragma unroll
  for (int c = 0; c < NRES; ++c)
    x0[c * XSTR + t] = w_in[c * MU + id] + b_in[c];
}

// ---------------------------------------------------------------------------
// One gated-residual block.  One wave = 16 time steps, 10 WMMAs:
//   s/t = Wsig/Wtanh(K=64 over two taps), g = sigmoid(s)*tanh(t),
//   res = Wres @ g ; xout = res + xin[:, t+d] ; G[block] <- g (f16, skip cols)
// ---------------------------------------------------------------------------
__global__ __launch_bounds__(256) void block_kernel(
    const float* __restrict__ xin, float* __restrict__ xout,
    const _Float16* __restrict__ Wg,       // 10 packed A fragments
    const float* __restrict__ bsig, const float* __restrict__ btanh,
    const float* __restrict__ bres,
    _Float16* __restrict__ Gout,           // [32][GSTR] for this block
    int d, int len_out, int off) {
  int wave = threadIdx.x >> 5;
  int l    = threadIdx.x & 31;
  int t0   = (blockIdx.x * 8 + wave) * 16;
  if (t0 >= len_out) return;               // wave-uniform exit
  int tmax = len_out - 1 + d;              // len_in - 1

  v16h B0 = load_bfrag_f32(xin, XSTR, t0, tmax);
  v16h B1 = load_bfrag_f32(xin, XSTR, t0 + d, tmax);
  const v16h* W = (const v16h*)Wg;
  v8f z = zero8();

  v8f slo = wmma_f16(W[0 * 32 + l], B0, z);
  slo     = wmma_f16(W[2 * 32 + l], B1, slo);
  v8f shi = wmma_f16(W[1 * 32 + l], B0, z);
  shi     = wmma_f16(W[3 * 32 + l], B1, shi);
  v8f tlo = wmma_f16(W[4 * 32 + l], B0, z);
  tlo     = wmma_f16(W[6 * 32 + l], B1, tlo);
  v8f thi = wmma_f16(W[5 * 32 + l], B0, z);
  thi     = wmma_f16(W[7 * 32 + l], B1, thi);

  int mb = (l >= 16) ? 8 : 0;
  v8f glo, ghi;
#pragma unroll
  for (int v = 0; v < 8; ++v) {
    glo[v] = fast_sigmoid(slo[v] + bsig[mb + v]) *
             fast_tanh(tlo[v] + btanh[mb + v]);
    ghi[v] = fast_sigmoid(shi[v] + bsig[16 + mb + v]) *
             fast_tanh(thi[v] + btanh[16 + mb + v]);
  }
  v16h Bg = dpair_to_bfrag(glo, ghi);

  v8f rlo = wmma_f16(W[8 * 32 + l], Bg, z);
  v8f rhi = wmma_f16(W[9 * 32 + l], Bg, z);

  int tcol = t0 + (l & 15);
  if (tcol < len_out) {
#pragma unroll
    for (int v = 0; v < 8; ++v) {
      int m0 = mb + v, m1 = 16 + mb + v;
      xout[m0 * XSTR + tcol] = rlo[v] + bres[m0] + xin[m0 * XSTR + tcol + d];
      xout[m1 * XSTR + tcol] = rhi[v] + bres[m1] + xin[m1 * XSTR + tcol + d];
    }
    int u = tcol - off;                    // skip-aligned column
    if (u >= 0) {
      int cb = (l >= 16) ? 16 : 0;
#pragma unroll
      for (int j = 0; j < 16; ++j)
        Gout[(cb + j) * GSTR + u] = Bg[j];
    }
  }
}

// ---------------------------------------------------------------------------
// Fused head: per 16-column tile, 256 threads (8 waves):
//  stage0: cooperatively stage the 50 G B-fragments into LDS (loaded once,
//          consumed by all 8 waves via ds_load_b128 pairs)
//  stage1: skip_sum = Wskip[512x1600] @ G  (wave w -> rows 64w..64w+63)
//  +Bsum, elu -> LDS B-frags; stage2: w_p1[512x512]; elu; stage3: w_p2[256x512]
//  Stage2/3 fragment buffer (16KB) aliases the G-fragment buffer (50KB).
// ---------------------------------------------------------------------------
__global__ __launch_bounds__(256) void head_kernel(
    const _Float16* __restrict__ G,
    const _Float16* __restrict__ WskipH, const _Float16* __restrict__ Wp1H,
    const _Float16* __restrict__ Wp2H,
    const float* __restrict__ Bsum, const float* __restrict__ bp1,
    const float* __restrict__ bp2, float* __restrict__ out) {
  __shared__ __align__(32) _Float16 smem[NBLOCKS * 32 * 16];  // 50KB
  int w  = threadIdx.x >> 5;
  int l  = threadIdx.x & 31;
  int u0 = blockIdx.x * 16;
  int mb = (l >= 16) ? 8 : 0;
  int cb = (l >= 16) ? 16 : 0;
  int u   = u0 + (l & 15);
  int ucl = (u < FINAL_LEN) ? u : (FINAL_LEN - 1);
  const v16h* Wskip = (const v16h*)WskipH;
  const v16h* Wp1   = (const v16h*)Wp1H;
  const v16h* Wp2   = (const v16h*)Wp2H;

  // ---- stage 0: stage all 50 G fragments into LDS (each loaded once) ----
  for (int kt = w; kt < NBLOCKS; kt += 8) {
    const _Float16* gp = G + ((size_t)kt * 32 + cb) * GSTR + ucl;
    v16h B;
#pragma unroll
    for (int j = 0; j < 16; ++j) B[j] = gp[(size_t)j * GSTR];
    *(v16h*)&smem[(kt * 32 + l) * 16] = B;
  }
  __syncthreads();

  // ---- stage 1: K = 1600 (kt == wavenet block id) ----
  v8f acc[4];
#pragma unroll
  for (int j = 0; j < 4; ++j) acc[j] = zero8();
  for (int kt = 0; kt < NBLOCKS; ++kt) {
    v16h B = *(const v16h*)&smem[(kt * 32 + l) * 16];
#pragma unroll
    for (int j = 0; j < 4; ++j) {
      int mt = w * 4 + j;
      acc[j] = wmma_f16(Wskip[(mt * NBLOCKS + kt) * 32 + l], B, acc[j]);
    }
  }
  __syncthreads();   // done reading G fragments; smem can be reused

  // +Bsum, elu, convert to B fragments in (aliased) LDS
#pragma unroll
  for (int p = 0; p < 2; ++p) {
    v8f lo = acc[2 * p], hi = acc[2 * p + 1];
    int mlo = (w * 4 + 2 * p) * 16;
#pragma unroll
    for (int v = 0; v < 8; ++v) {
      lo[v] = elu(lo[v] + Bsum[mlo + mb + v]);
      hi[v] = elu(hi[v] + Bsum[mlo + 16 + mb + v]);
    }
    *(v16h*)&smem[((w * 2 + p) * 32 + l) * 16] = dpair_to_bfrag(lo, hi);
  }
  __syncthreads();

  // ---- stage 2: h1 = w_p1 @ elu(skip_sum), K = 512 (16 k-steps) ----
  v8f acc2[4];
#pragma unroll
  for (int j = 0; j < 4; ++j) acc2[j] = zero8();
  for (int kt = 0; kt < 16; ++kt) {
    v16h B = *(const v16h*)&smem[(kt * 32 + l) * 16];
#pragma unroll
    for (int j = 0; j < 4; ++j)
      acc2[j] = wmma_f16(Wp1[((w * 4 + j) * 16 + kt) * 32 + l], B, acc2[j]);
  }
  __syncthreads();   // everyone done reading stage-2 fragments
#pragma unroll
  for (int p = 0; p < 2; ++p) {
    v8f lo = acc2[2 * p], hi = acc2[2 * p + 1];
    int mlo = (w * 4 + 2 * p) * 16;
#pragma unroll
    for (int v = 0; v < 8; ++v) {
      lo[v] = elu(lo[v] + bp1[mlo + mb + v]);
      hi[v] = elu(hi[v] + bp1[mlo + 16 + mb + v]);
    }
    *(v16h*)&smem[((w * 2 + p) * 32 + l) * 16] = dpair_to_bfrag(lo, hi);
  }
  __syncthreads();

  // ---- stage 3: out = w_p2 @ elu(h1), M = 256 (wave w -> mtiles 2w,2w+1) ----
  v8f acc3[2];
#pragma unroll
  for (int j = 0; j < 2; ++j) acc3[j] = zero8();
  for (int kt = 0; kt < 16; ++kt) {
    v16h B = *(const v16h*)&smem[(kt * 32 + l) * 16];
#pragma unroll
    for (int j = 0; j < 2; ++j)
      acc3[j] = wmma_f16(Wp2[((w * 2 + j) * 16 + kt) * 32 + l], B, acc3[j]);
  }
  if (u < FINAL_LEN) {
#pragma unroll
    for (int j = 0; j < 2; ++j)
#pragma unroll
      for (int v = 0; v < 8; ++v) {
        int m = (w * 2 + j) * 16 + mb + v;
        out[(size_t)u * MU + m] = acc3[j][v] + bp2[m];
      }
  }
}

// ---------------------------------------------------------------------------
extern "C" void kernel_launch(void* const* d_in, const int* in_sizes, int n_in,
                              void* d_out, int out_size, void* d_ws, size_t ws_size,
                              hipStream_t stream) {
  (void)in_sizes; (void)n_in; (void)out_size; (void)ws_size;
  const int*   idx    = (const int*)d_in[0];
  const float* w_in   = (const float*)d_in[1];
  const float* b_in   = (const float*)d_in[2];
  const float* w_sig  = (const float*)d_in[3];
  const float* b_sig  = (const float*)d_in[4];
  const float* w_tanh = (const float*)d_in[5];
  const float* b_tanh = (const float*)d_in[6];
  const float* w_skip = (const float*)d_in[7];
  const float* b_skip = (const float*)d_in[8];
  const float* w_res  = (const float*)d_in[9];
  const float* b_res  = (const float*)d_in[10];
  const float* w_p1   = (const float*)d_in[11];
  const float* b_p1   = (const float*)d_in[12];
  const float* w_p2   = (const float*)d_in[13];
  const float* b_p2   = (const float*)d_in[14];
  float* out = (float*)d_out;

  char* ws = (char*)d_ws;
  size_t off = 0;
  auto alloc = [&](size_t bytes) -> void* {
    void* p = ws + off;
    off = (off + bytes + 255) & ~(size_t)255;
    return p;
  };
  float*    x0    = (float*)alloc((size_t)NRES * XSTR * 4);
  float*    x1    = (float*)alloc((size_t)NRES * XSTR * 4);
  _Float16* G     = (_Float16*)alloc((size_t)NBLOCKS * NRES * GSTR * 2);
  _Float16* Wskip = (_Float16*)alloc((size_t)NSK_FRAGS * 512 * 2);
  _Float16* Wp1   = (_Float16*)alloc((size_t)NP1_FRAGS * 512 * 2);
  _Float16* Wp2   = (_Float16*)alloc((size_t)NP2_FRAGS * 512 * 2);
  _Float16* Wgate = (_Float16*)alloc((size_t)NGT_FRAGS * 512 * 2);
  float*    Bsum  = (float*)alloc(NSKIP * 4);

  pack_kernel<<<(TOTAL_FRAGS * 32 + 127) / 128, 128, 0, stream>>>(
      w_sig, w_tanh, w_skip, w_res, w_p1, w_p2, Wskip, Wp1, Wp2, Wgate);
  bsum_kernel<<<2, 256, 0, stream>>>(b_skip, Bsum);
  embed_kernel<<<(T_IN + 255) / 256, 256, 0, stream>>>(idx, w_in, b_in, x0);

  int len = T_IN;
  float* xa = x0;
  float* xb = x1;
  for (int i = 0; i < NBLOCKS; ++i) {
    int d = 1 << (i % 10);
    int len_out = len - d;
    int offc = len_out - FINAL_LEN;
    int ntiles = (len_out + 15) / 16;
    block_kernel<<<(ntiles + 7) / 8, 256, 0, stream>>>(
        xa, xb, Wgate + (size_t)i * 10 * 512,
        b_sig + i * NRES, b_tanh + i * NRES, b_res + i * NRES,
        G + (size_t)i * NRES * GSTR, d, len_out, offc);
    float* tmp = xa; xa = xb; xb = tmp;
    len = len_out;
  }

  head_kernel<<<(FINAL_LEN + 15) / 16, 256, 0, stream>>>(
      G, Wskip, Wp1, Wp2, Bsum, b_p1, b_p2, out);
}
